// MpoxMamba_23708219473993
// MI455X (gfx1250) — matrix operand order repaired and checked by reading, not compiled, not run on hardware
//
#include <hip/hip_runtime.h>

#define EPSF 1e-5f

typedef __attribute__((ext_vector_type(16))) _Float16 v16h;
typedef __attribute__((ext_vector_type(8)))  float    v8f;

// ---------------------------------------------------------------- activations
template<int ACT>
__device__ __forceinline__ float actf(float v){
  if (ACT==1) return fminf(fmaxf(v, 0.f), 6.f);          // relu6
  if (ACT==2) return v * (1.f / (1.f + __expf(-v)));     // silu
  return v;
}

// ------------------------------------------------ CDNA5 async global->LDS copy
// VDST operand = LDS byte address; ISA 10.2: LDS aperture uses addr[31:0], so
// truncating the generic pointer to 32 bits yields the hardware LDS address.
__device__ __forceinline__ void async_g2l_b128(const void* lds, const void* g){
  unsigned l = (unsigned)(unsigned long long)lds;
  asm volatile("global_load_async_to_lds_b128 %0, %1, off"
               :: "v"(l), "v"((unsigned long long)g) : "memory");
}
__device__ __forceinline__ void wait_async0(){
#if __has_builtin(__builtin_amdgcn_s_wait_asynccnt)
  __builtin_amdgcn_s_wait_asynccnt(0);
#else
  asm volatile("s_wait_asynccnt 0" ::: "memory");
#endif
}

// row r of a 16-row tile -> (batch, spatial) without per-element division:
// base_b/base_r come from ONE 32-bit division per block; a tile crosses at
// most one batch-row boundary because HW >= 16 for every layer.
__device__ __forceinline__ void rowdec(int base_b, int base_r, int r, int HW,
                                       int& bb, int& rr){
  rr = base_r + r;
  bb = base_b;
  if (rr >= HW) { rr -= HW; bb += 1; }
}

// ---------------------------------------------------------------- WMMA GEMM
// C[M,N] = epi( A[M,K] * W^T ),  W given row-major (N,K)  (i.e. B[k][n]=W[n*K+k])
// AMODE/OMODE: 0 = row-major (lda/ldc + col offset), 1 = NCHW channels-major:
//   addr = batch*batchStride + k*chanStride + spatial + off
// For AMODE/OMODE 1, M = B*HW is a multiple of 16 (all layers), so no row
// clamping is needed. LDS tile layouts: A: AMODE0 -> [r][k] (k contiguous),
// AMODE1 -> [k][r] (m contiguous, aHW % 4 == 0 for all layers); B: [n][k].
// Full interior tiles use GLOBAL_LOAD_ASYNC_TO_LDS_B128 (ASYNCcnt); ragged
// tiles use branch-free clamped loads + ds stores.
template<int AMODE,int OMODE,int ACT,int HASBN,int HASBIAS,int HASRES>
__global__ void __launch_bounds__(32)
gemm16(const float* __restrict__ A, const float* __restrict__ Wt,
       float* __restrict__ O, const float* __restrict__ R,
       const float* __restrict__ bng, const float* __restrict__ bnb,
       const float* __restrict__ bnm, const float* __restrict__ bnv,
       const float* __restrict__ bias,
       int M, int N, int K,
       long aLda, long aOff, long aBatch, long aChan, int aHW,
       long oLdc, long oOff, long oBatch, long oChan, int oHW)
{
  __shared__ float aLDS[16 * 32];
  __shared__ float bLDS[16 * 32];
  const int lane = threadIdx.x;            // wave32: 0..31
  const int tm = blockIdx.x * 16;
  const int tn = blockIdx.y * 16;
  const bool fullMN = (tm + 16 <= M) && (tn + 16 <= N);

  int ab0 = 0, ar0 = 0;                    // single division per block (AMODE 1)
  if (AMODE == 1) { ab0 = tm / aHW; ar0 = tm - ab0 * aHW; }

  v8f acc = {};
  for (int k0 = 0; k0 < K; k0 += 32) {
    if (k0 + 32 < K)                       // near prefetch of next weight tile
      __builtin_prefetch(&Wt[(long)tn * K + k0 + 32], 0, 3);

    if (fullMN && (k0 + 32 <= K)) {
      // ---------------- fast path: async DMA tiles into LDS ----------------
      #pragma unroll
      for (int i = 0; i < 4; ++i) {                 // A tile: 128 x b128
        int q = lane + 32 * i;                      // 0..127
        const float* gp; const float* lp;
        if (AMODE == 0) {
          int r = q >> 3, c4 = (q & 7) * 4;
          gp = A + (long)(tm + r) * aLda + (k0 + c4) + aOff;
          lp = aLDS + (r * 32 + c4);
        } else {
          int kk = q >> 2, r4 = (q & 3) * 4;
          int bb, rr; rowdec(ab0, ar0, r4, aHW, bb, rr);
          gp = A + (long)bb * aBatch + (long)(k0 + kk) * aChan + rr + aOff;
          lp = aLDS + (kk * 16 + r4);
        }
        async_g2l_b128(lp, gp);
      }
      #pragma unroll
      for (int i = 0; i < 4; ++i) {                 // B tile: 128 x b128
        int q = lane + 32 * i;
        int nn = q >> 3, c4 = (q & 7) * 4;
        async_g2l_b128(bLDS + (nn * 32 + c4), Wt + (long)(tn + nn) * K + (k0 + c4));
      }
      wait_async0();
    } else {
      // ------------- edge path: branch-free clamped loads ------------------
      #pragma unroll
      for (int i = 0; i < 16; ++i) {
        int idx = i * 32 + lane;
        int r = idx >> 5, c = idx & 31;
        int mm = tm + r, kk = k0 + c;
        int kc = kk < K - 1 ? kk : K - 1;
        long addr; bool mOK;
        if (AMODE == 0) {
          int mc = mm < M - 1 ? mm : M - 1;
          addr = (long)mc * aLda + kc + aOff;
          mOK = (mm < M);
        } else {                                    // M multiple of 16 here
          int bb, rr; rowdec(ab0, ar0, r, aHW, bb, rr);
          addr = (long)bb * aBatch + (long)kc * aChan + rr + aOff;
          mOK = true;
        }
        float v = A[addr];
        if (!mOK || kk >= K) v = 0.f;
        aLDS[(AMODE == 0) ? (r * 32 + c) : (c * 16 + r)] = v;
      }
      #pragma unroll
      for (int i = 0; i < 16; ++i) {
        int idx = i * 32 + lane;
        int kk = idx >> 4, nn = idx & 15;
        int kg = k0 + kk, ng = tn + nn;
        int kc = kg < K - 1 ? kg : K - 1;
        int nc = ng < N - 1 ? ng : N - 1;
        float v = Wt[(long)nc * K + kc];
        if (kg >= K || ng >= N) v = 0.f;
        bLDS[nn * 32 + kk] = v;
      }
    }

    // fragment assembly per CDNA5 16-bit WMMA layouts (f32 -> f16 on read);
    // wave-private LDS, compiler inserts s_wait_dscnt before the WMMA.
    const int half = lane >> 4;
    const int mr   = lane & 15;
    v16h af, bf;
    #pragma unroll
    for (int e = 0; e < 16; ++e) {
      int p = e >> 1, o = e & 1;
      int ka = (p < 4 ? 2 * p : 16 + 2 * (p - 4)) + 8 * half + o;  // A 16x32
      float av = (AMODE == 0) ? aLDS[mr * 32 + ka] : aLDS[ka * 16 + mr];
      af[e] = (_Float16)av;
      int kb = (e < 8 ? e : e + 8) + 8 * half;                     // B 32x16
      bf[e] = (_Float16)bLDS[mr * 32 + kb];
    }
    acc = __builtin_amdgcn_wmma_f32_16x16x32_f16(
        false, af, false, bf, (short)0, acc, false, false);
  }

  // epilogue: C layout -> lane: n = lane&15 ; vgpr r: m = r + 8*(lane>>4)
  const int n = tn + (lane & 15);
  const int mOffHi = (lane >> 4) ? 8 : 0;
  const int mb = tm + mOffHi;
  int ob0 = 0, or0 = 0;
  if (OMODE == 1) { ob0 = tm / oHW; or0 = tm - ob0 * oHW; }
  if (n < N) {
    float scale = 1.f, shift = 0.f, bs = 0.f;
    if (HASBN) {
      float g = bng[n], b = bnb[n], mu = bnm[n], vv = bnv[n];
      scale = g * rsqrtf(vv + EPSF);
      shift = b - mu * scale;
    }
    if (HASBIAS) bs = bias[n];
    #pragma unroll
    for (int r = 0; r < 8; ++r) {
      int mm = mb + r;
      if (mm < M) {
        float v = acc[r] + bs;
        if (HASBN) v = v * scale + shift;
        v = actf<ACT>(v);
        long addr;
        if (OMODE == 0) {
          addr = (long)mm * oLdc + n + oOff;
        } else {
          int bb, rr; rowdec(ob0, or0, mOffHi + r, oHW, bb, rr);
          addr = (long)bb * oBatch + (long)n * oChan + rr + oOff;
        }
        if (HASRES) v += R[addr];
        O[addr] = v;
      }
    }
  }
}

// ---------------------------------------------------------------- stem conv
__global__ void conv_b1(const float* __restrict__ X, const float* __restrict__ Wt,
                        const float* __restrict__ bias, float* __restrict__ Y,
                        int B, int H, int W, int OH, int OW)
{
  int idx = blockIdx.x * blockDim.x + threadIdx.x;
  int total = B * 32 * OH * OW;
  if (idx >= total) return;
  int ow = idx % OW; int t = idx / OW;
  int oh = t % OH;   t /= OH;
  int co = t % 32;   int b = t / 32;
  float acc = bias[co];
  for (int ci = 0; ci < 3; ++ci)
    for (int ky = 0; ky < 3; ++ky) {
      int iy = oh * 2 - 1 + ky;
      if (iy < 0 || iy >= H) continue;
      for (int kx = 0; kx < 3; ++kx) {
        int ix = ow * 2 - 1 + kx;
        if (ix < 0 || ix >= W) continue;
        acc += X[((long)(b * 3 + ci) * H + iy) * W + ix] *
               Wt[((co * 3 + ci) * 3 + ky) * 3 + kx];
      }
    }
  Y[idx] = acc;
}

// ---------------------------------------------------------------- depthwise 3x3 + BN + act
template<int ACT,int HASBIAS>
__global__ void dwconv3x3_bn(const float* __restrict__ X, const float* __restrict__ Wt,
                             const float* __restrict__ bias,
                             const float* __restrict__ bng, const float* __restrict__ bnb,
                             const float* __restrict__ bnm, const float* __restrict__ bnv,
                             float* __restrict__ Y,
                             int B, int C, int H, int W, int OH, int OW, int stride)
{
  int idx = blockIdx.x * blockDim.x + threadIdx.x;
  int total = B * C * OH * OW;
  if (idx >= total) return;
  int ow = idx % OW; int t = idx / OW;
  int oh = t % OH;   t /= OH;
  int ch = t % C;    int b = t / C;
  float acc = HASBIAS ? bias[ch] : 0.f;
  const float* xp = X + (long)(b * C + ch) * H * W;
  const float* wp = Wt + ch * 9;
  for (int ky = 0; ky < 3; ++ky) {
    int iy = oh * stride - 1 + ky;
    if (iy < 0 || iy >= H) continue;
    for (int kx = 0; kx < 3; ++kx) {
      int ix = ow * stride - 1 + kx;
      if (ix < 0 || ix >= W) continue;
      acc += xp[iy * W + ix] * wp[ky * 3 + kx];
    }
  }
  float s = bng[ch] * rsqrtf(bnv[ch] + EPSF);
  acc = (acc - bnm[ch]) * s + bnb[ch];
  Y[idx] = actf<ACT>(acc);
}

// ------------------------------------------------ SS2D depthwise conv (BHWC in, BCL out, silu)
__global__ void ss2d_dwconv(const float* __restrict__ xz, const float* __restrict__ Wt,
                            const float* __restrict__ bias, float* __restrict__ xc,
                            int B, int di, int H, int W)
{
  int L = H * W;
  int idx = blockIdx.x * blockDim.x + threadIdx.x;
  int total = B * di * L;
  if (idx >= total) return;
  int l = idx % L; int t = idx / L;
  int d = t % di;  int b = t / di;
  int h = l / W, w = l % W;
  float acc = bias[d];
  const float* wp = Wt + d * 9;
  for (int ky = 0; ky < 3; ++ky) {
    int iy = h - 1 + ky;
    if (iy < 0 || iy >= H) continue;
    for (int kx = 0; kx < 3; ++kx) {
      int ix = w - 1 + kx;
      if (ix < 0 || ix >= W) continue;
      acc += xz[((long)b * L + iy * W + ix) * (2 * di) + d] * wp[ky * 3 + kx];
    }
  }
  xc[idx] = actf<2>(acc);
}

// ---------------------------------------------------------------- cross scan (4 directions)
__global__ void cross_scan(const float* __restrict__ xc, float* __restrict__ xs,
                           int B, int di, int H, int W)
{
  int L = H * W;
  int idx = blockIdx.x * blockDim.x + threadIdx.x;
  int total = B * 4 * di * L;
  if (idx >= total) return;
  int l = idx % L; int t = idx / L;
  int d = t % di;  t /= di;
  int k = t % 4;   int b = t / 4;
  int src;
  if (k == 0)      src = l;
  else if (k == 1) { int w = l / H, h = l % H; src = h * W + w; }
  else if (k == 2) src = L - 1 - l;
  else { int l2 = L - 1 - l; int w = l2 / H, h = l2 % H; src = h * W + w; }
  xs[idx] = xc[(long)(b * di + d) * L + src];
}

// ---------------------------------------------------------------- dt projection + softplus
__global__ void dt_softplus(const float* __restrict__ xdbl, const float* __restrict__ dtw,
                            const float* __restrict__ dtb, float* __restrict__ dts,
                            int B, int di, int dr, int L)
{
  int Cp = dr + 32;
  int idx = blockIdx.x * blockDim.x + threadIdx.x;
  int total = B * 4 * di * L;
  if (idx >= total) return;
  int l = idx % L; int t = idx / L;
  int d = t % di;  t /= di;
  int k = t % 4;   int b = t / 4;
  float s = dtb[k * di + d];
  for (int r = 0; r < dr; ++r)
    s += xdbl[((long)(b * 4 + k) * Cp + r) * L + l] * dtw[(k * di + d) * dr + r];
  dts[idx] = (s > 20.f) ? s : __logf(1.f + __expf(s));
}

// ---------------------------------------------------------------- selective scan (sequential in L)
__global__ void ss_scan(const float* __restrict__ xs, const float* __restrict__ dts,
                        const float* __restrict__ xdbl, const float* __restrict__ Alog,
                        const float* __restrict__ Dp, float* __restrict__ ys,
                        int B, int di, int dr, int L)
{
  int idx = blockIdx.x * blockDim.x + threadIdx.x;
  int total = B * 4 * di;
  if (idx >= total) return;
  int d = idx % di; int t = idx / di;
  int k = t % 4;    int b = t / 4;
  int Cp = dr + 32;
  const float* Bs = xdbl + ((long)(b * 4 + k) * Cp + dr) * L;
  const float* Cs = Bs + (long)16 * L;
  const float* u  = xs  + ((long)(b * 4 + k) * di + d) * L;
  const float* dt = dts + ((long)(b * 4 + k) * di + d) * L;
  float*       yo = ys  + ((long)(b * 4 + k) * di + d) * L;
  float Dv = Dp[k * di + d];
  float Aa[16], h[16];
  #pragma unroll
  for (int n = 0; n < 16; ++n) { Aa[n] = -__expf(Alog[(k * di + d) * 16 + n]); h[n] = 0.f; }
  for (int l = 0; l < L; ++l) {
    float uv = u[l], dtv = dt[l], y = 0.f;
    #pragma unroll
    for (int n = 0; n < 16; ++n) {
      h[n] = h[n] * __expf(dtv * Aa[n]) + dtv * uv * Bs[(long)n * L + l];
      y += h[n] * Cs[(long)n * L + l];
    }
    yo[l] = y + Dv * uv;
  }
}

// ---------------------------------------------------------------- merge 4 directions -> (B,L,di)
__global__ void ss_merge(const float* __restrict__ ys, float* __restrict__ ym,
                         int B, int di, int H, int W)
{
  int L = H * W;
  int idx = blockIdx.x * blockDim.x + threadIdx.x;
  int total = B * L * di;
  if (idx >= total) return;
  int d = idx % di; int t = idx / di;
  int l = t % L;    int b = t / L;
  int h = l / W, w = l % W;
  long bd = (long)b * 4 * di * L + (long)d * L;
  long dl = (long)di * L;
  int wh = w * H + h;
  float v = ys[bd + l]                       // direction 0
          + ys[bd + 2 * dl + (L - 1 - l)]    // direction 2 flipped
          + ys[bd + 1 * dl + wh]             // direction 1 via wh
          + ys[bd + 3 * dl + (L - 1 - wh)];  // direction 3 flipped via wh
  ym[idx] = v;
}

// ---------------------------------------------------------------- LayerNorm * silu(z)
__global__ void ln_gate(const float* __restrict__ ym, const float* __restrict__ xz,
                        const float* __restrict__ g, const float* __restrict__ bb,
                        float* __restrict__ ya, int BL, int di)
{
  int row = blockIdx.x * blockDim.x + threadIdx.x;
  if (row >= BL) return;
  const float* y = ym + (long)row * di;
  float mu = 0.f;
  for (int d = 0; d < di; ++d) mu += y[d];
  mu /= di;
  float var = 0.f;
  for (int d = 0; d < di; ++d) { float t = y[d] - mu; var += t * t; }
  var /= di;
  float inv = rsqrtf(var + EPSF);
  const float* z = xz + (long)row * 2 * di + di;
  float* o = ya + (long)row * di;
  for (int d = 0; d < di; ++d) {
    float v = (y[d] - mu) * inv * g[d] + bb[d];
    float zz = z[d];
    o[d] = v * (zz / (1.f + __expf(-zz)));
  }
}

// ---------------------------------------------------------------- global average pool (B,C,HW)->(B,C)
__global__ void gap_kernel(const float* __restrict__ X, float* __restrict__ g,
                           int B, int C, int HW)
{
  int idx = blockIdx.x * blockDim.x + threadIdx.x;
  if (idx >= B * C) return;
  const float* p = X + (long)idx * HW;
  float s = 0.f;
  for (int i = 0; i < HW; ++i) s += p[i];
  g[idx] = s / HW;
}

// ---------------------------------------------------------------- ECA gate -> concat[:, :C]
__global__ void eca_apply(const float* __restrict__ X, const float* __restrict__ g,
                          const float* __restrict__ w3, float* __restrict__ concatB,
                          int B, int C, int HW)
{
  int idx = blockIdx.x * blockDim.x + threadIdx.x;
  int total = B * C * HW;
  if (idx >= total) return;
  int hw = idx % HW; int t = idx / HW;
  int c = t % C;     int b = t / C;
  float gm = (c > 0)     ? g[b * C + c - 1] : 0.f;
  float g0 = g[b * C + c];
  float gp = (c < C - 1) ? g[b * C + c + 1] : 0.f;
  float a = w3[0] * gm + w3[1] * g0 + w3[2] * gp;
  a = 1.f / (1.f + __expf(-a));
  concatB[((long)b * 2 * C + c) * HW + hw] = X[((long)b * C + c) * HW + hw] * a;
}

// ---------------------------------------------------- silu(transpose(sbuf)) -> concat[:, C:2C]
__global__ void s_to_concat(const float* __restrict__ sbuf, float* __restrict__ concatB,
                            int B, int C, int HW)
{
  int idx = blockIdx.x * blockDim.x + threadIdx.x;
  int total = B * C * HW;
  if (idx >= total) return;
  int hw = idx % HW; int t = idx / HW;
  int c = t % C;     int b = t / C;
  float v = sbuf[((long)b * HW + hw) * C + c];
  concatB[((long)b * 2 * C + C + c) * HW + hw] = actf<2>(v);
}

// ================================================================ host side
struct BNP   { const float *g, *b, *m, *v; };
struct IRP   { const float* pw1w; BNP bn1; const float* dww; BNP bn2; const float* pw2w; BNP bn3; };
struct SS2DP { const float *inproj, *convw, *convb, *xproj, *dtw, *dtb, *Alog, *D, *lng, *lnb, *outproj; };
struct GSSMP { const float *dww, *dwb; BNP bn1; const float *pww, *pwb; BNP bn2;
               SS2DP ssm[4]; const float *finw, *finb, *ecaw; };

static inline dim3 gg(int M, int N) { return dim3((M + 15) / 16, (N + 15) / 16); }
static inline int  nb(int total)    { return (total + 255) / 256; }

static void ss2d_run(hipStream_t st, const SS2DP& q,
                     const float* srcCM, int Ctot, int chOff,
                     int m, int B, int H, int W,
                     float* sbuf, int sldc, int sColOff, float* scr)
{
  const int L = H * W, BL = B * L, di = 2 * m, dr = (m + 15) / 16, Cp = dr + 32;
  float* xz   = scr;
  float* xc   = xz   + (size_t)BL * 2 * di;
  float* xs   = xc   + (size_t)BL * di;
  float* xdbl = xs   + (size_t)BL * 4 * di;
  float* dts  = xdbl + (size_t)BL * 4 * Cp;
  float* ys   = dts  + (size_t)BL * 4 * di;
  float* ym   = ys   + (size_t)BL * 4 * di;
  float* ya   = ym   + (size_t)BL * di;

  // in_proj: (B*L, m) x (m, 2di) -> xz row-major
  gemm16<1,0,0,0,0,0><<<gg(BL, 2 * di), 32, 0, st>>>(
      srcCM, q.inproj, xz, nullptr, nullptr, nullptr, nullptr, nullptr, nullptr,
      BL, 2 * di, m,
      0, (long)chOff * L, (long)Ctot * L, L, L,
      (long)(2 * di), 0, 0, 0, 1);

  ss2d_dwconv<<<nb(B * di * L), 256, 0, st>>>(xz, q.convw, q.convb, xc, B, di, H, W);
  cross_scan<<<nb(B * 4 * di * L), 256, 0, st>>>(xc, xs, B, di, H, W);

  // x_proj per direction: (B*L, di) x (di, Cp) -> xdbl channels-major
  for (int k = 0; k < 4; ++k)
    gemm16<1,1,0,0,0,0><<<gg(BL, Cp), 32, 0, st>>>(
        xs, q.xproj + (size_t)k * Cp * di, xdbl, nullptr,
        nullptr, nullptr, nullptr, nullptr, nullptr,
        BL, Cp, di,
        0, (long)k * di * L, (long)4 * di * L, L, L,
        0, (long)k * Cp * L, (long)4 * Cp * L, L, L);

  dt_softplus<<<nb(B * 4 * di * L), 256, 0, st>>>(xdbl, q.dtw, q.dtb, dts, B, di, dr, L);
  ss_scan<<<(B * 4 * di + 63) / 64, 64, 0, st>>>(xs, dts, xdbl, q.Alog, q.D, ys, B, di, dr, L);
  ss_merge<<<nb(B * L * di), 256, 0, st>>>(ys, ym, B, di, H, W);
  ln_gate<<<(BL + 127) / 128, 128, 0, st>>>(ym, xz, q.lng, q.lnb, ya, BL, di);

  // out_proj: (B*L, di) x (di, m) -> sbuf (row-major, col offset)
  gemm16<0,0,0,0,0,0><<<gg(BL, m), 32, 0, st>>>(
      ya, q.outproj, sbuf, nullptr, nullptr, nullptr, nullptr, nullptr, nullptr,
      BL, m, di,
      (long)di, 0, 0, 0, 1,
      (long)sldc, (long)sColOff, 0, 0, 1);
}

static void gssm_run(hipStream_t st, const GSSMP& p, int c, int B, int H, int W,
                     const float* X, float* Y, float* t1, float* outb, float* ssmR)
{
  const int HW = H * W, BHW = B * HW, m = c / 4;
  dwconv3x3_bn<2,1><<<nb(B * c * HW), 256, 0, st>>>(
      X, p.dww, p.dwb, p.bn1.g, p.bn1.b, p.bn1.m, p.bn1.v, t1, B, c, H, W, H, W, 1);
  gemm16<1,1,2,1,1,0><<<gg(BHW, c), 32, 0, st>>>(
      t1, p.pww, outb, nullptr, p.bn2.g, p.bn2.b, p.bn2.m, p.bn2.v, p.pwb,
      BHW, c, c,
      0, 0, (long)c * HW, HW, HW,
      0, 0, (long)c * HW, HW, HW);

  float* g    = ssmR;
  float* sbuf = ssmR + 2048;
  float* scr  = sbuf + (size_t)BHW * c;
  float* concatB = t1;                                   // t1 dead, reuse as concat

  gap_kernel<<<nb(B * c), 256, 0, st>>>(outb, g, B, c, HW);
  eca_apply<<<nb(B * c * HW), 256, 0, st>>>(outb, g, p.ecaw, concatB, B, c, HW);

  for (int i = 0; i < 4; ++i)
    ss2d_run(st, p.ssm[i], outb, c, i * m, m, B, H, W, sbuf, c, i * m, scr);

  s_to_concat<<<nb(B * c * HW), 256, 0, st>>>(sbuf, concatB, B, c, HW);

  // final 1x1: (B*HW, 2c) x (2c, c) + bias, silu, + residual X
  gemm16<1,1,2,0,1,1><<<gg(BHW, c), 32, 0, st>>>(
      concatB, p.finw, Y, X, nullptr, nullptr, nullptr, nullptr, p.finb,
      BHW, c, 2 * c,
      0, 0, (long)2 * c * HW, HW, HW,
      0, 0, (long)c * HW, HW, HW);
}

static void ir_run(hipStream_t st, const IRP& p, int Ci, int Co, int B, int H, int W,
                   int stride, bool shortcut, const float* X, float* Y, float* t1, float* t2)
{
  const int h = 2 * Ci, HW = H * W;
  const int OH = (H - 1) / stride + 1, OW = (W - 1) / stride + 1, OHW = OH * OW;
  gemm16<1,1,1,1,0,0><<<gg(B * HW, h), 32, 0, st>>>(
      X, p.pw1w, t1, nullptr, p.bn1.g, p.bn1.b, p.bn1.m, p.bn1.v, nullptr,
      B * HW, h, Ci,
      0, 0, (long)Ci * HW, HW, HW,
      0, 0, (long)h * HW, HW, HW);
  dwconv3x3_bn<1,0><<<nb(B * h * OHW), 256, 0, st>>>(
      t1, p.dww, nullptr, p.bn2.g, p.bn2.b, p.bn2.m, p.bn2.v, t2, B, h, H, W, OH, OW, stride);
  if (shortcut)
    gemm16<1,1,0,1,0,1><<<gg(B * OHW, Co), 32, 0, st>>>(
        t2, p.pw2w, Y, X, p.bn3.g, p.bn3.b, p.bn3.m, p.bn3.v, nullptr,
        B * OHW, Co, h,
        0, 0, (long)h * OHW, OHW, OHW,
        0, 0, (long)Co * OHW, OHW, OHW);
  else
    gemm16<1,1,0,1,0,0><<<gg(B * OHW, Co), 32, 0, st>>>(
        t2, p.pw2w, Y, nullptr, p.bn3.g, p.bn3.b, p.bn3.m, p.bn3.v, nullptr,
        B * OHW, Co, h,
        0, 0, (long)h * OHW, OHW, OHW,
        0, 0, (long)Co * OHW, OHW, OHW);
}

extern "C" void kernel_launch(void* const* d_in, const int* in_sizes, int n_in,
                              void* d_out, int out_size, void* d_ws, size_t ws_size,
                              hipStream_t stream)
{
  (void)in_sizes; (void)n_in; (void)out_size; (void)ws_size;
  int ii = 0;
  auto nx  = [&]() -> const float* { return (const float*)d_in[ii++]; };
  auto nbn = [&]() { BNP b; b.g = nx(); b.b = nx(); b.m = nx(); b.v = nx(); return b; };
  auto nir = [&]() { IRP r; r.pw1w = nx(); r.bn1 = nbn(); r.dww = nx(); r.bn2 = nbn();
                     r.pw2w = nx(); r.bn3 = nbn(); return r; };
  auto nss = [&]() { SS2DP s; s.inproj = nx(); s.convw = nx(); s.convb = nx();
                     s.xproj = nx(); s.dtw = nx(); s.dtb = nx(); s.Alog = nx();
                     s.D = nx(); s.lng = nx(); s.lnb = nx(); s.outproj = nx(); return s; };
  auto ngs = [&]() { GSSMP g; g.dww = nx(); g.dwb = nx(); g.bn1 = nbn();
                     g.pww = nx(); g.pwb = nx(); g.bn2 = nbn();
                     for (int i = 0; i < 4; ++i) g.ssm[i] = nss();
                     g.finw = nx(); g.finb = nx(); g.ecaw = nx(); return g; };

  const float* x   = nx();
  const float* b1w = nx(); const float* b1b = nx();
  IRP   b2  = nir(); IRP   b3  = nir();
  GSSMP b4  = ngs(); GSSMP b5  = ngs();
  IRP   b6  = nir();
  GSSMP b7  = ngs(); GSSMP b8  = ngs(); GSSMP b9 = ngs();
  IRP   b10 = nir();
  const float* finw = nx(); BNP finbn = nbn();
  const float* clsw = nx(); const float* clsb = nx();

  const int B = 8;
  char* wsb = (char*)d_ws;
  float* bufA = (float*)(wsb);                      // 32 MB ping
  float* bufB = (float*)(wsb + (32ull << 20));      // 32 MB pong
  float* bufC = (float*)(wsb + (64ull << 20));      // 32 MB scratch (t1 / concat)
  float* bufD = (float*)(wsb + (96ull << 20));      // 32 MB scratch (t2 / pw-out)
  float* ssmR = (float*)(wsb + (128ull << 20));     // SS2D region (reused per branch)

  // stem: 3 -> 32, stride 2, pad 1
  conv_b1<<<nb(B * 32 * 112 * 112), 256, 0, stream>>>(x, b1w, b1b, bufA, B, 224, 224, 112, 112);
  const float* cur = bufA;

  ir_run(stream, b2, 32, 32, B, 112, 112, 1, true,  cur, bufB, bufC, bufD); cur = bufB;
  ir_run(stream, b3, 32, 64, B, 112, 112, 2, false, cur, bufA, bufC, bufD); cur = bufA;

  gssm_run(stream, b4, 64, B, 56, 56, cur, bufB, bufC, bufD, ssmR); cur = bufB;
  gssm_run(stream, b5, 64, B, 56, 56, cur, bufA, bufC, bufD, ssmR); cur = bufA;

  ir_run(stream, b6, 64, 128, B, 56, 56, 2, false, cur, bufB, bufC, bufD); cur = bufB;

  gssm_run(stream, b7, 128, B, 28, 28, cur, bufA, bufC, bufD, ssmR); cur = bufA;
  gssm_run(stream, b8, 128, B, 28, 28, cur, bufB, bufC, bufD, ssmR); cur = bufB;
  gssm_run(stream, b9, 128, B, 28, 28, cur, bufA, bufC, bufD, ssmR); cur = bufA;

  ir_run(stream, b10, 128, 256, B, 28, 28, 2, false, cur, bufB, bufC, bufD); cur = bufB;

  // head: 1x1 conv 256 -> 512 + BN + silu at 14x14
  const int HWh = 14 * 14;
  gemm16<1,1,2,1,0,0><<<gg(B * HWh, 512), 32, 0, stream>>>(
      cur, finw, bufC, nullptr, finbn.g, finbn.b, finbn.m, finbn.v, nullptr,
      B * HWh, 512, 256,
      0, 0, (long)256 * HWh, HWh, HWh,
      0, 0, (long)512 * HWh, HWh, HWh);
  gap_kernel<<<nb(B * 512), 256, 0, stream>>>(bufC, bufD, B, 512, HWh);

  // classifier: (8,512) x (512,10) + bias -> d_out
  gemm16<0,0,0,0,1,0><<<gg(B, 10), 32, 0, stream>>>(
      bufD, clsw, (float*)d_out, nullptr, nullptr, nullptr, nullptr, nullptr, clsb,
      B, 10, 512,
      (long)512, 0, 0, 0, 1,
      (long)10, 0, 0, 0, 1);
}
